// ViT_7834020348624
// MI455X (gfx1250) — matrix operand order repaired and checked by reading, not compiled, not run on hardware
//
#include <hip/hip_runtime.h>
#include <hip/hip_bf16.h>

// ---------------- ViT config (matches reference) ----------------
#define BATCH    64
#define SEQL     197      // 14*14 + 1
#define NPATCH   196
#define DHID     768
#define NHEADS   12
#define DHEAD    64
#define NBLOCKS  12
#define OUTD     1000
#define OUTP     1008     // padded logits stride
#define INPUTD   768      // 3*16*16
#define MROWS    (BATCH*NPATCH)   // 12544

// attention tiling (strides chosen as multiples of 8 bf16 = 16B alignment)
#define SP       224      // padded seq (14 tiles of 16)
#define QROWS    112      // per-workgroup query rows (7 tiles)
#define LDQ      72       // row stride for sQ / sK (bf16 elements)
#define LDVT     224      // row stride for transposed V (t dimension)
#define LDP      224      // row stride for probabilities

typedef __attribute__((ext_vector_type(16))) __bf16 bf16x16;
typedef __attribute__((ext_vector_type(8)))  __bf16 bf16x8;
typedef __attribute__((ext_vector_type(8)))  float  f32x8;
typedef __attribute__((ext_vector_type(4)))  int    i32x4;

#define AS1 __attribute__((address_space(1)))
#define AS3 __attribute__((address_space(3)))

// ---------------- CDNA5 async global->LDS staging ------------------------
#if __has_builtin(__builtin_amdgcn_global_load_async_to_lds_b128)
#define HAVE_ASYNC_LDS 1
#endif

// copy 16 bytes (8 bf16) global -> LDS
__device__ __forceinline__ void stage16(__bf16* ldst, const __bf16* gsrc) {
#if defined(HAVE_ASYNC_LDS)
  __builtin_amdgcn_global_load_async_to_lds_b128((i32x4*)gsrc,
                                                 (AS3 i32x4*)ldst, 0, 0);
#else
  *(bf16x8*)ldst = *(const bf16x8*)gsrc;
#endif
}

__device__ __forceinline__ void stage_wait() {
#if defined(HAVE_ASYNC_LDS)
#if __has_builtin(__builtin_amdgcn_s_wait_asynccnt)
  __builtin_amdgcn_s_wait_asynccnt(0);
#else
  asm volatile("s_wait_asynccnt 0" ::: "memory");
#endif
#endif
}

// ---------------- WMMA helpers (wave32, bf16 -> f32 accum) ----------------
__device__ __forceinline__ f32x8 wmma_bf16(bf16x16 a, bf16x16 b, f32x8 c) {
  return __builtin_amdgcn_wmma_f32_16x16x32_bf16(false, a, false, b,
                                                 (short)0, c, false, false);
}

__device__ __forceinline__ f32x8 zero8() {
  f32x8 z;
#pragma unroll
  for (int v = 0; v < 8; ++v) z[v] = 0.0f;
  return z;
}

__device__ __forceinline__ bf16x16 combine(bf16x8 lo, bf16x8 hi) {
  bf16x16 r;
#pragma unroll
  for (int v = 0; v < 8; ++v) { r[v] = lo[v]; r[v + 8] = hi[v]; }
  return r;
}

// packed bf16 pair store (one b32 store, uses v_cvt_pk_bf16_f32)
__device__ __forceinline__ void store_pk2(__bf16* dst, float a, float b) {
  union { unsigned u; __bf16 h[2]; } pk;
  pk.h[0] = (__bf16)a;
  pk.h[1] = (__bf16)b;
  *(unsigned*)dst = pk.u;
}

// A-matrix 16x32 fragment from row-major tile (origin applied). lane[3:0]=M,
// lane[4] selects K half (+8). Elements k..k+7 / k+16..k+23 are contiguous
// -> two 16B vector loads (ds_load_b128). ld must be a multiple of 8.
__device__ __forceinline__ bf16x16 load_a_frag(const __bf16* base, int ld) {
  const int lane = threadIdx.x & 31;
  const __bf16* p = base + (lane & 15) * ld + ((lane >> 4) ? 8 : 0);
  return combine(*(const bf16x8*)p, *(const bf16x8*)(p + 16));
}

// B-matrix 32x16 fragment from N-MAJOR (transposed) storage bt[n][k].
// lane[3:0]=N col, lane[4] selects K half (+8): k..k+7 / k+16..k+23
// -> two 16B vector loads. ld and k0 must be multiples of 8.
__device__ __forceinline__ bf16x16 load_b_frag_t(const __bf16* bt, int ld,
                                                 int n0, int k0) {
  const int lane = threadIdx.x & 31;
  const __bf16* p =
      bt + (size_t)(n0 + (lane & 15)) * ld + k0 + (lane >> 4) * 8;
  return combine(*(const bf16x8*)p, *(const bf16x8*)(p + 16));
}

// A fragment from global activations x[b, row, col0..] with row<SEQL guard.
__device__ __forceinline__ bf16x16 load_a_x(const __bf16* x, int b, int row0,
                                            int col0) {
  const int lane = threadIdx.x & 31;
  const int m = row0 + (lane & 15);
  if (m < SEQL) {
    const __bf16* p = x + ((size_t)b * SEQL + m) * DHID + col0 +
                      ((lane >> 4) ? 8 : 0);
    return combine(*(const bf16x8*)p, *(const bf16x8*)(p + 16));
  }
  bf16x16 z;
#pragma unroll
  for (int v = 0; v < 16; ++v) z[v] = (__bf16)0.0f;
  return z;
}

// positional embedding (used only by the small table kernel)
__device__ __forceinline__ float pos_emb(int s, int d) {
  const float angle =
      (float)s * __powf(10000.0f, -((float)d) * (1.0f / (float)DHID));
  return (d & 1) ? __cosf(angle) : __sinf(angle);
}

// ---------------- K0: positional-embedding table (197 x 768 f32) ----------
__global__ __launch_bounds__(256) void pos_table_kernel(
    float* __restrict__ pos) {
  const int idx = blockIdx.x * 256 + threadIdx.x;
  if (idx >= SEQL * DHID) return;
  const int s = idx / DHID, d = idx % DHID;
  pos[idx] = pos_emb(s, d);
}

// ---------------- K1: patchify images -> bf16 A matrix (12544 x 768) ------
__global__ __launch_bounds__(256) void patchify_kernel(
    const float* __restrict__ images, __bf16* __restrict__ A) {
  const long long idx = (long long)blockIdx.x * 256 + threadIdx.x;
  const long long total = (long long)MROWS * INPUTD;
  if (idx >= total) return;
  const int j = (int)(idx % INPUTD);
  const long long row = idx / INPUTD;
  const int b = (int)(row / NPATCH);
  const int p = (int)(row % NPATCH);
  const int c = j >> 8;
  const int jj = j & 255;
  const int dy = jj >> 4, dx = jj & 15;
  const int py = p / 14, px = p % 14;
  const float v =
      images[(((size_t)b * 3 + c) * 224 + (py * 16 + dy)) * 224 + px * 16 + dx];
  A[idx] = (__bf16)v;
}

// ---------------- K2: class token row (s = 0) -----------------------------
__global__ __launch_bounds__(256) void cls_kernel(
    const float* __restrict__ v_class, const float* __restrict__ pos,
    __bf16* __restrict__ x) {
  const int idx = blockIdx.x * 256 + threadIdx.x;
  if (idx >= BATCH * DHID) return;
  const int b = idx / DHID, d = idx % DHID;
  x[(size_t)b * SEQL * DHID + d] = (__bf16)(v_class[d] + pos[d]);
}

// ---------------- K3: embed GEMM, fused bias + positional -----------------
// C[12544 x 768] = A @ W_emb ; block computes 128x64, 8 waves x (16 x 64)
__global__ __launch_bounds__(256) void embed_gemm_kernel(
    const __bf16* __restrict__ A, const float* __restrict__ W,
    const float* __restrict__ bias, const float* __restrict__ pos,
    __bf16* __restrict__ xout) {
  __shared__ __bf16 sA[128 * 40];    // row-major A tile
  __shared__ __bf16 sBT[64 * 32];    // N-major (transposed) B tile
  const int tid = threadIdx.x, wave = tid >> 5, lane = tid & 31;
  const int m0 = blockIdx.x * 128, n0 = blockIdx.y * 64;
  f32x8 acc[4];
#pragma unroll
  for (int t = 0; t < 4; ++t) acc[t] = zero8();

  for (int kk = 0; kk < INPUTD; kk += 32) {
    __syncthreads();
    // A tile: 128x32, async global->LDS (512 chunks of 8 bf16)
#pragma unroll
    for (int i = 0; i < 2; ++i) {
      const int li = tid * 2 + i;
      const int r = li >> 2, c = (li & 3) * 8;
      stage16(&sA[r * 40 + c], &A[(size_t)(m0 + r) * INPUTD + kk + c]);
    }
    // B tile 32x64 -> transposed sBT[n][k], packed-pair bf16 stores
#pragma unroll
    for (int i = 0; i < 4; ++i) {
      const int li = tid * 4 + i;          // pair index (1024 pairs)
      const int n = li & 63, k = (li >> 6) * 2;
      const float w0 = W[(size_t)(kk + k) * DHID + n0 + n];
      const float w1 = W[(size_t)(kk + k + 1) * DHID + n0 + n];
      store_pk2(&sBT[n * 32 + k], w0, w1);
    }
    if (kk + 32 < INPUTD)  // CDNA5 global_prefetch_b8 of the next A slice
      __builtin_prefetch(&A[(size_t)(m0 + (tid >> 1)) * INPUTD + kk + 32], 0, 1);
    stage_wait();
    __syncthreads();

    const bf16x16 af = load_a_frag(sA + wave * 16 * 40, 40);
#pragma unroll
    for (int t = 0; t < 4; ++t)
      acc[t] = wmma_bf16(af, load_b_frag_t(sBT, 32, t * 16, 0), acc[t]);
  }

  const int nl = lane & 15, mb = (lane >> 4) * 8;
#pragma unroll
  for (int t = 0; t < 4; ++t) {
    const int n = n0 + t * 16 + nl;
    const float bn = bias[n];
#pragma unroll
    for (int v = 0; v < 8; ++v) {
      const int m = m0 + wave * 16 + mb + v;
      const int b = m / NPATCH, s = (m % NPATCH) + 1;
      const float val = acc[t][v] + bn + pos[s * DHID + n];
      xout[((size_t)b * SEQL + s) * DHID + n] = (__bf16)val;
    }
  }
}

// ---------------- K4: one attention layer, LDS-resident per (b,h,half) ----
// grid = BATCH*NHEADS*2 ; 256 threads = 8 waves; ~149 KB static LDS
__global__ __launch_bounds__(256) void attn_kernel(
    const __bf16* __restrict__ xin, __bf16* __restrict__ xout,
    const float* __restrict__ Wq, const float* __restrict__ Wk,
    const float* __restrict__ Wv, const float* __restrict__ bq,
    const float* __restrict__ bk, const float* __restrict__ bv) {
  __shared__ __bf16 sQ[QROWS * LDQ];    // my half of Q, row-major [r][e]
  __shared__ __bf16 sK[SP * LDQ];       // full K, row-major [t][e] (=N-major B)
  __shared__ __bf16 sVT[DHEAD * LDVT];  // full V, TRANSPOSED [e][t]
  __shared__ __bf16 sP[QROWS * LDP];    // scores -> probabilities [r][t]
  __shared__ __bf16 sWT[3 * DHEAD * DHEAD];  // Wq/Wk/Wv transposed [e][d]
  __shared__ float  sBias[3 * DHEAD];

  const int tid = threadIdx.x, wave = tid >> 5, lane = tid & 31;
  const int idx = blockIdx.x;
  const int half = idx & 1;
  const int bh = idx >> 1;
  const int b = bh / NHEADS, h = bh % NHEADS;
  const int qbase = half * QROWS;
  const int nl = lane & 15, mb = (lane >> 4) * 8;

  // stage weights (f32 -> bf16, transposed, packed pairs) + biases into LDS
  for (int li = tid; li < 6144; li += 256) {   // 6144 bf16 pairs
    const int wsel = li >> 11;                 // 0=Q 1=K 2=V
    const int r = li & 2047;
    const int e = r >> 5;                      // 32 pairs per row
    const int d = (r & 31) * 2;                // sWT[e][d] = W[d][e]
    const float* src = (wsel == 0) ? Wq : ((wsel == 1) ? Wk : Wv);
    const float w0 = src[(size_t)h * 4096 + d * 64 + e];
    const float w1 = src[(size_t)h * 4096 + (d + 1) * 64 + e];
    store_pk2(&sWT[wsel * 4096 + e * 64 + d], w0, w1);
  }
  if (tid < 3 * DHEAD) {
    const float* src = (tid < 64) ? bq : ((tid < 128) ? bk : bv);
    sBias[tid] = src[h * DHEAD + (tid & 63)];
  }
  __syncthreads();

  // ---- Q/K/V projections: 84 jobs (K+V fused: 56, Q: 28) ----
  for (int job = wave; job < 84; job += 8) {
    if (job < 56) {  // K and V share the X fragments
      const int i = job >> 2, j = job & 3, r0 = i * 16;
      const bf16x16 a0 = load_a_x(xin, b, r0, h * DHEAD);
      const bf16x16 a1 = load_a_x(xin, b, r0, h * DHEAD + 32);
      f32x8 ak = zero8(), av = zero8();
      ak = wmma_bf16(a0, load_b_frag_t(sWT + 4096, 64, j * 16, 0), ak);
      ak = wmma_bf16(a1, load_b_frag_t(sWT + 4096, 64, j * 16, 32), ak);
      av = wmma_bf16(a0, load_b_frag_t(sWT + 8192, 64, j * 16, 0), av);
      av = wmma_bf16(a1, load_b_frag_t(sWT + 8192, 64, j * 16, 32), av);
      const float bik = sBias[64 + j * 16 + nl];
      const float biv = sBias[128 + j * 16 + nl];
#pragma unroll
      for (int v = 0; v < 8; ++v)  // K row-major (scalar column stores)
        sK[(r0 + mb + v) * LDQ + j * 16 + nl] = (__bf16)(ak[v] + bik);
      bf16x8 vv;                   // V transposed: 8 contiguous t -> 16B store
#pragma unroll
      for (int v = 0; v < 8; ++v) vv[v] = (__bf16)(av[v] + biv);
      *(bf16x8*)(&sVT[(size_t)(j * 16 + nl) * LDVT + r0 + mb]) = vv;
    } else {  // Q for my half rows
      const int jj = job - 56;
      const int i = jj >> 2, j = jj & 3;
      const int r0 = qbase + i * 16;
      f32x8 acc = zero8();
      acc = wmma_bf16(load_a_x(xin, b, r0, h * DHEAD),
                      load_b_frag_t(sWT, 64, j * 16, 0), acc);
      acc = wmma_bf16(load_a_x(xin, b, r0, h * DHEAD + 32),
                      load_b_frag_t(sWT, 64, j * 16, 32), acc);
      const float bi = sBias[j * 16 + nl];
#pragma unroll
      for (int v = 0; v < 8; ++v)
        sQ[(i * 16 + mb + v) * LDQ + j * 16 + nl] = (__bf16)(acc[v] + bi);
    }
  }
  __syncthreads();

  // ---- scores = Q @ K^T * 1/sqrt(64) : 7 x 14 tiles ----
  // B operand is K^T; row-major sK[t][e] is exactly N-major storage for it.
  for (int job = wave; job < 98; job += 8) {
    const int i = job / 14, j = job % 14;
    f32x8 acc = zero8();
#pragma unroll
    for (int k0 = 0; k0 < DHEAD; k0 += 32) {
      const bf16x16 af = load_a_frag(sQ + i * 16 * LDQ + k0, LDQ);
      const bf16x16 bf_ = load_b_frag_t(sK, LDQ, j * 16, k0);
      acc = wmma_bf16(af, bf_, acc);
    }
#pragma unroll
    for (int v = 0; v < 8; ++v)
      sP[(i * 16 + mb + v) * LDP + j * 16 + nl] = (__bf16)(acc[v] * 0.125f);
  }
  __syncthreads();

  // ---- row softmax with sequence mask (cols >= 197 -> 0), vectorized ----
  if (tid < QROWS) {
    __bf16* row = sP + tid * LDP;
    const int s = qbase + tid;
    if (s < SEQL) {
      float mx = -1e30f;
      for (int c = 0; c < 200; c += 8) {
        const bf16x8 v8 = *(const bf16x8*)(row + c);
#pragma unroll
        for (int u = 0; u < 8; ++u)
          if (c + u < SEQL) mx = fmaxf(mx, (float)v8[u]);
      }
      float sum = 0.0f;
      for (int c = 0; c < 200; c += 8) {
        const bf16x8 v8 = *(const bf16x8*)(row + c);
#pragma unroll
        for (int u = 0; u < 8; ++u)
          if (c + u < SEQL) sum += __expf((float)v8[u] - mx);
      }
      const float inv = 1.0f / sum;
      for (int c = 0; c < SP; c += 8) {
        const bf16x8 v8 = *(const bf16x8*)(row + c);
        bf16x8 o8;
#pragma unroll
        for (int u = 0; u < 8; ++u) {
          const int t = c + u;
          o8[u] = (t < SEQL) ? (__bf16)(__expf((float)v8[u] - mx) * inv)
                             : (__bf16)0.0f;
        }
        *(bf16x8*)(row + c) = o8;
      }
    } else {
      bf16x8 z8;
#pragma unroll
      for (int u = 0; u < 8; ++u) z8[u] = (__bf16)0.0f;
      for (int c = 0; c < SP; c += 8) *(bf16x8*)(row + c) = z8;
    }
  }
  __syncthreads();

  // ---- out = P @ V : 7 x 4 tiles, K = 224 (V consumed transposed) ----
  for (int job = wave; job < 28; job += 8) {
    const int i = job >> 2, j = job & 3;
    f32x8 acc = zero8();
#pragma unroll
    for (int ks = 0; ks < 7; ++ks) {
      const bf16x16 af = load_a_frag(sP + i * 16 * LDP + ks * 32, LDP);
      const bf16x16 bf_ = load_b_frag_t(sVT, LDVT, j * 16, ks * 32);
      acc = wmma_bf16(af, bf_, acc);
    }
#pragma unroll
    for (int v = 0; v < 8; ++v) {
      const int s = qbase + i * 16 + mb + v;
      if (s < SEQL)
        xout[((size_t)b * SEQL + s) * DHID + h * DHEAD + j * 16 + nl] =
            (__bf16)acc[v];
    }
  }
}

// ---------------- K5: head GEMM (64 x 768) @ (768 x 1000) + bias ----------
__global__ __launch_bounds__(128) void head_gemm_kernel(
    const __bf16* __restrict__ x, const float* __restrict__ Wh,
    const float* __restrict__ bh, float* __restrict__ logits) {
  __shared__ __bf16 sBT[64 * 32];  // transposed B tile [n][k]
  const int tid = threadIdx.x, wave = tid >> 5, lane = tid & 31;
  const int n0 = blockIdx.x * 64;
  f32x8 acc[4];
#pragma unroll
  for (int t = 0; t < 4; ++t) acc[t] = zero8();

  for (int kk = 0; kk < DHID; kk += 32) {
    __syncthreads();
    // 32x64 B tile, transposed, packed pairs, guard n < 1000
#pragma unroll
    for (int i = 0; i < 8; ++i) {
      const int li = tid * 8 + i;          // pair index (1024 pairs)
      const int n = li & 63, k = (li >> 6) * 2;
      const int ng = n0 + n;
      const float w0 = (ng < OUTD) ? Wh[(size_t)(kk + k) * OUTD + ng] : 0.0f;
      const float w1 =
          (ng < OUTD) ? Wh[(size_t)(kk + k + 1) * OUTD + ng] : 0.0f;
      store_pk2(&sBT[n * 32 + k], w0, w1);
    }
    __syncthreads();
    // A fragment straight from global cls rows (row stride SEQL*DHID)
    const __bf16* p = x + (size_t)(wave * 16 + (lane & 15)) * SEQL * DHID + kk +
                      ((lane >> 4) ? 8 : 0);
    const bf16x16 af = combine(*(const bf16x8*)p, *(const bf16x8*)(p + 16));
#pragma unroll
    for (int t = 0; t < 4; ++t)
      acc[t] = wmma_bf16(af, load_b_frag_t(sBT, 32, t * 16, 0), acc[t]);
  }

  const int nl = lane & 15, mb = (lane >> 4) * 8;
#pragma unroll
  for (int t = 0; t < 4; ++t) {
    const int n = n0 + t * 16 + nl;
    if (n >= OUTD) continue;
#pragma unroll
    for (int v = 0; v < 8; ++v) {
      const int m = wave * 16 + mb + v;
      logits[(size_t)m * OUTP + n] = acc[t][v] + bh[n];
    }
  }
}

// ---------------- K6: row softmax over 1000 classes -----------------------
__global__ __launch_bounds__(256) void softmax_kernel(
    const float* __restrict__ logits, float* __restrict__ out) {
  __shared__ float red[256];
  const int row = blockIdx.x, tid = threadIdx.x;
  const float* lr = logits + (size_t)row * OUTP;

  float mx = -1e30f;
  for (int n = tid; n < OUTD; n += 256) mx = fmaxf(mx, lr[n]);
  red[tid] = mx;
  __syncthreads();
  for (int s = 128; s > 0; s >>= 1) {
    if (tid < s) red[tid] = fmaxf(red[tid], red[tid + s]);
    __syncthreads();
  }
  mx = red[0];
  __syncthreads();

  float sum = 0.0f;
  for (int n = tid; n < OUTD; n += 256) sum += __expf(lr[n] - mx);
  red[tid] = sum;
  __syncthreads();
  for (int s = 128; s > 0; s >>= 1) {
    if (tid < s) red[tid] += red[tid + s];
    __syncthreads();
  }
  sum = red[0];

  const float inv = 1.0f / sum;
  for (int n = tid; n < OUTD; n += 256)
    out[(size_t)row * OUTD + n] = __expf(lr[n] - mx) * inv;
}

// ---------------- launcher ------------------------------------------------
extern "C" void kernel_launch(void* const* d_in, const int* in_sizes, int n_in,
                              void* d_out, int out_size, void* d_ws,
                              size_t ws_size, hipStream_t stream) {
  const float* images  = (const float*)d_in[0];
  // d_in[1] = n_patches scalar (constexpr here)
  const float* W_emb   = (const float*)d_in[2];
  const float* b_emb   = (const float*)d_in[3];
  const float* v_class = (const float*)d_in[4];
  const float* Wq      = (const float*)d_in[5];
  const float* Wk      = (const float*)d_in[6];
  const float* Wv      = (const float*)d_in[7];
  const float* bq      = (const float*)d_in[8];
  const float* bk      = (const float*)d_in[9];
  const float* bv      = (const float*)d_in[10];
  const float* W_head  = (const float*)d_in[11];
  const float* b_head  = (const float*)d_in[12];
  (void)in_sizes; (void)n_in; (void)out_size; (void)ws_size;

  // workspace layout (bytes, all 256-aligned)
  char* ws = (char*)d_ws;
  const size_t szA   = (size_t)MROWS * INPUTD * 2;          // 19,267,584
  const size_t szX   = (size_t)BATCH * SEQL * DHID * 2;     // 19,365,888
  const size_t szLog = (size_t)BATCH * OUTP * 4;            // 258,048
  __bf16* A      = (__bf16*)(ws);
  __bf16* xa     = (__bf16*)(ws + szA);
  __bf16* xb     = (__bf16*)(ws + szA + szX);
  float*  logits = (float*) (ws + szA + 2 * szX);
  float*  pos    = (float*) (ws + szA + 2 * szX + szLog);   // 197 x 768 f32

  // 0) positional-embedding table (transcendentals computed once)
  pos_table_kernel<<<(SEQL * DHID + 255) / 256, 256, 0, stream>>>(pos);
  // 1) patchify -> bf16 A matrix
  {
    const long long total = (long long)MROWS * INPUTD;
    const int blocks = (int)((total + 255) / 256);
    patchify_kernel<<<blocks, 256, 0, stream>>>(images, A);
  }
  // 2) class token rows
  cls_kernel<<<(BATCH * DHID + 255) / 256, 256, 0, stream>>>(v_class, pos, xa);
  // 3) embedding GEMM with fused bias + positional epilogue
  embed_gemm_kernel<<<dim3(MROWS / 128, DHID / 64), 256, 0, stream>>>(
      A, W_emb, b_emb, pos, xa);
  // 4) 12 attention-only transformer blocks (ping-pong xa/xb)
  for (int l = 0; l < NBLOCKS; ++l) {
    const __bf16* xin = (l & 1) ? xb : xa;
    __bf16* xout      = (l & 1) ? xa : xb;
    attn_kernel<<<BATCH * NHEADS * 2, 256, 0, stream>>>(
        xin, xout,
        Wq + (size_t)l * NHEADS * DHEAD * DHEAD,
        Wk + (size_t)l * NHEADS * DHEAD * DHEAD,
        Wv + (size_t)l * NHEADS * DHEAD * DHEAD,
        bq + (size_t)l * NHEADS * DHEAD,
        bk + (size_t)l * NHEADS * DHEAD,
        bv + (size_t)l * NHEADS * DHEAD);
  }
  // 5) classification head GEMM (reads cls rows of xa)
  head_gemm_kernel<<<16, 128, 0, stream>>>(xa, W_head, b_head, logits);
  // 6) softmax over classes
  softmax_kernel<<<BATCH, 256, 0, stream>>>(logits, (float*)d_out);
}
